// MixtureOfExperts_38860864094324
// MI455X (gfx1250) — matrix operand order repaired and checked by reading, not compiled, not run on hardware
//
#include <hip/hip_runtime.h>

#define N_TOK 16384
#define DIM   1024
#define NE    64
#define TOPK  2
#define CAP   640

typedef __attribute__((ext_vector_type(16))) __bf16 v16bf;
typedef __attribute__((ext_vector_type(8)))  float  v8f;
typedef __attribute__((ext_vector_type(4)))  float  f32x4;
typedef __attribute__((ext_vector_type(4)))  unsigned u32x4;

union FragBF { v16bf v; u32x4 q[2]; };

static __device__ __forceinline__ unsigned short f2bf(float f) {
    unsigned u = __builtin_bit_cast(unsigned, f);
    unsigned r = u + 0x7FFFu + ((u >> 16) & 1u);   // round-to-nearest-even
    return (unsigned short)(r >> 16);
}
static __device__ __forceinline__ unsigned pack2(float lo, float hi) {
    return (unsigned)f2bf(lo) | ((unsigned)f2bf(hi) << 16);
}

#define LDA 40   // padded halves per row: 20 dwords -> conflict-free strides
#define LDB 40

// ---------------------------------------------------------------------------
// Kernel 1: routing logits = x[N,D] @ Wr[D,E] in bf16 WMMA, fp32 accum
// Tile: 128 (M) x 64 (N=E), K-step 32. 8 waves: wave w owns rows w*16..w*16+15.
// ---------------------------------------------------------------------------
__global__ __launch_bounds__(256) void k_routing_gemm(const float* __restrict__ x,
                                                      const float* __restrict__ Wr,
                                                      float* __restrict__ logits) {
    __shared__ __align__(16) unsigned short As[128 * LDA];
    __shared__ __align__(16) unsigned short Bs[64 * LDB];
    const int tid = threadIdx.x;
    const int m0 = blockIdx.x * 128;
    const int lane = tid & 31, w = tid >> 5;
    const int hi = lane >> 4, ln = lane & 15;

    v8f acc[4] = {};

    for (int k0 = 0; k0 < DIM; k0 += 32) {
        // stage A tile (128 rows x 32 k) as bf16
        for (int i = 0; i < 4; ++i) {
            int idx = tid + i * 256;
            int r = idx >> 3, c4 = idx & 7;
            f32x4 f = *(const f32x4*)(x + (size_t)(m0 + r) * DIM + k0 + c4 * 4);
            unsigned* dst = (unsigned*)&As[r * LDA + c4 * 4];
            dst[0] = pack2(f[0], f[1]);
            dst[1] = pack2(f[2], f[3]);
        }
        // stage B tile (32 k x 64 n) transposed -> Bs[n][k]
        {
            int kr2 = tid >> 4, n4 = tid & 15;
            f32x4 fa = *(const f32x4*)(Wr + (size_t)(k0 + kr2 * 2) * NE + n4 * 4);
            f32x4 fb = *(const f32x4*)(Wr + (size_t)(k0 + kr2 * 2 + 1) * NE + n4 * 4);
            for (int j = 0; j < 4; ++j)
                *(unsigned*)&Bs[(n4 * 4 + j) * LDB + kr2 * 2] = pack2(fa[j], fb[j]);
        }
        __syncthreads();

        FragBF fA;
        int row = w * 16 + ln;
        fA.q[0] = *(const u32x4*)&As[row * LDA + hi * 8];
        fA.q[1] = *(const u32x4*)&As[row * LDA + 16 + hi * 8];
        for (int nt = 0; nt < 4; ++nt) {
            FragBF fB;
            int col = nt * 16 + ln;
            fB.q[0] = *(const u32x4*)&Bs[col * LDB + hi * 8];
            fB.q[1] = *(const u32x4*)&Bs[col * LDB + 16 + hi * 8];
            acc[nt] = __builtin_amdgcn_wmma_f32_16x16x32_bf16(
                false, fA.v, false, fB.v, (short)0, acc[nt], false, false);
        }
        __syncthreads();
    }
    for (int nt = 0; nt < 4; ++nt)
        for (int v = 0; v < 8; ++v) {
            int row = m0 + w * 16 + v + hi * 8;
            int col = nt * 16 + ln;
            logits[(size_t)row * NE + col] = acc[nt][v];
        }
}

// ---------------------------------------------------------------------------
// Kernel 2: per-token top-2 + softmax gates
// ---------------------------------------------------------------------------
__global__ void k_top2(const float* __restrict__ logits, const float* __restrict__ br,
                       int* __restrict__ topk, float* __restrict__ gates) {
    int n = blockIdx.x * blockDim.x + threadIdx.x;
    if (n >= N_TOK) return;
    const float* row = logits + (size_t)n * NE;
    float v0 = -3.4e38f, v1 = -3.4e38f;
    int i0 = 0, i1 = 0;
    for (int e = 0; e < NE; ++e) {
        float v = row[e] + br[e];
        if (v > v0) { v1 = v0; i1 = i0; v0 = v; i0 = e; }
        else if (v > v1) { v1 = v; i1 = e; }
    }
    float e1 = __expf(v1 - v0);
    float inv = 1.0f / (1.0f + e1);
    topk[2 * n] = i0;  topk[2 * n + 1] = i1;
    gates[2 * n] = inv; gates[2 * n + 1] = e1 * inv;
}

// ---------------------------------------------------------------------------
// Kernel 3: init slot->token map to -1 (fresh every call)
// ---------------------------------------------------------------------------
__global__ void k_init_slots(int* __restrict__ slot_token) {
    int i = blockIdx.x * 256 + threadIdx.x;
    if (i < NE * CAP) slot_token[i] = -1;
}

// ---------------------------------------------------------------------------
// Kernel 4: deterministic capacity positions. One block per expert scans the
// k-major flattened assignment list (flat[f] with f = kk*N + n), exactly
// matching the reference cumsum order so the same tokens overflow.
// ---------------------------------------------------------------------------
__global__ __launch_bounds__(256) void k_positions(const int* __restrict__ topk,
                                                   int* __restrict__ position,
                                                   int* __restrict__ slot_token,
                                                   int* __restrict__ counts) {
    const int e = blockIdx.x;
    const int tid = threadIdx.x;
    __shared__ int sa[256], sb[256];
    int running = 0;
    for (int f0 = 0; f0 < TOPK * N_TOK; f0 += 256) {
        int f = f0 + tid;
        int n = f & (N_TOK - 1);
        int kk = f >> 14;                       // N_TOK == 2^14
        int m = (topk[2 * n + kk] == e) ? 1 : 0;
        sa[tid] = m;
        __syncthreads();
        int* a = sa; int* b = sb;
        for (int off = 1; off < 256; off <<= 1) {
            int v = a[tid];
            if (tid >= off) v += a[tid - off];
            b[tid] = v;
            __syncthreads();
            int* t = a; a = b; b = t;
        }
        int incl = a[tid];
        int total = a[255];
        if (m) {
            int pos = running + incl - 1;       // exclusive count == cumsum-1
            position[f] = pos;
            if (pos < CAP) slot_token[e * CAP + pos] = n;
        }
        running += total;
        __syncthreads();
    }
    if (tid == 0) counts[e] = running < CAP ? running : CAP;
}

// ---------------------------------------------------------------------------
// Kernel 5: grouped expert GEMM: relu(gather(x) @ We[e]) -> expert_out[e,c,h]
// Grid (8 nblk, 5 mblk, 64 experts); 128x128 tile; 8 waves as 4(M)x2(N);
// each wave: 2 m-subtiles x 4 n-subtiles = 8 WMMA accum tiles.
// ---------------------------------------------------------------------------
__global__ __launch_bounds__(256) void k_expert_gemm(const float* __restrict__ x,
                                                     const float* __restrict__ We,
                                                     const int* __restrict__ slot_token,
                                                     const int* __restrict__ counts,
                                                     float* __restrict__ eout) {
    const int e = blockIdx.z;
    const int m0 = blockIdx.y * 128;
    const int n0 = blockIdx.x * 128;
    if (m0 >= counts[e]) return;                // skip empty capacity blocks

    __shared__ __align__(16) unsigned short As[128 * LDA];
    __shared__ __align__(16) unsigned short Bs[128 * LDB];
    __shared__ int toks[128];

    const int tid = threadIdx.x;
    if (tid < 128) toks[tid] = slot_token[e * CAP + m0 + tid];
    __syncthreads();

    const int lane = tid & 31, w = tid >> 5;
    const int hi = lane >> 4, ln = lane & 15;
    const int mw = w >> 1, nw = w & 1;
    const float* Wb = We + (size_t)e * DIM * DIM;

    v8f acc[2][4] = {};

    for (int k0 = 0; k0 < DIM; k0 += 32) {
        // stage A: gathered token rows, bf16, zero rows for empty slots
        for (int i = 0; i < 4; ++i) {
            int idx = tid + i * 256;
            int r = idx >> 3, c4 = idx & 7;
            int tok = toks[r];
            f32x4 f = {0.f, 0.f, 0.f, 0.f};
            if (tok >= 0) f = *(const f32x4*)(x + (size_t)tok * DIM + k0 + c4 * 4);
            unsigned* dst = (unsigned*)&As[r * LDA + c4 * 4];
            dst[0] = pack2(f[0], f[1]);
            dst[1] = pack2(f[2], f[3]);
        }
        // stage B: We tile (32 k x 128 n) transposed -> Bs[n][k]
        for (int i = 0; i < 2; ++i) {
            int idx = tid + i * 256;
            int kr2 = idx >> 5, n4 = idx & 31;
            f32x4 fa = *(const f32x4*)(Wb + (size_t)(k0 + kr2 * 2) * DIM + n0 + n4 * 4);
            f32x4 fb = *(const f32x4*)(Wb + (size_t)(k0 + kr2 * 2 + 1) * DIM + n0 + n4 * 4);
            for (int j = 0; j < 4; ++j)
                *(unsigned*)&Bs[(n4 * 4 + j) * LDB + kr2 * 2] = pack2(fa[j], fb[j]);
        }
        __syncthreads();

        FragBF fA[2], fB[4];
        for (int mt = 0; mt < 2; ++mt) {
            int row = (mw * 2 + mt) * 16 + ln;
            fA[mt].q[0] = *(const u32x4*)&As[row * LDA + hi * 8];
            fA[mt].q[1] = *(const u32x4*)&As[row * LDA + 16 + hi * 8];
        }
        for (int nt = 0; nt < 4; ++nt) {
            int col = (nw * 4 + nt) * 16 + ln;
            fB[nt].q[0] = *(const u32x4*)&Bs[col * LDB + hi * 8];
            fB[nt].q[1] = *(const u32x4*)&Bs[col * LDB + 16 + hi * 8];
        }
        for (int mt = 0; mt < 2; ++mt)
            for (int nt = 0; nt < 4; ++nt)
                acc[mt][nt] = __builtin_amdgcn_wmma_f32_16x16x32_bf16(
                    false, fA[mt].v, false, fB[nt].v, (short)0, acc[mt][nt], false, false);
        __syncthreads();
    }

    for (int mt = 0; mt < 2; ++mt)
        for (int nt = 0; nt < 4; ++nt)
            for (int v = 0; v < 8; ++v) {
                int c = m0 + (mw * 2 + mt) * 16 + v + hi * 8;
                int h = n0 + (nw * 4 + nt) * 16 + ln;
                float val = acc[mt][nt][v];
                eout[((size_t)e * CAP + c) * DIM + h] = val > 0.f ? val : 0.f;
            }
}

// ---------------------------------------------------------------------------
// Kernel 6: combine: out[n] = g0*eout[e0,p0] + g1*eout[e1,p1]; dropped -> 0
// ---------------------------------------------------------------------------
__global__ __launch_bounds__(256) void k_combine(const float* __restrict__ eout,
                                                 const int* __restrict__ topk,
                                                 const float* __restrict__ gates,
                                                 const int* __restrict__ position,
                                                 float* __restrict__ out) {
    const int n = blockIdx.x;
    const int h = threadIdx.x;                   // float4 index, 256*4 = 1024
    int e0 = topk[2 * n], e1 = topk[2 * n + 1];
    float g0 = gates[2 * n], g1 = gates[2 * n + 1];
    int p0 = position[n];
    int p1 = position[N_TOK + n];
    f32x4 a = {0.f, 0.f, 0.f, 0.f}, b = {0.f, 0.f, 0.f, 0.f};
    if (p0 < CAP) a = ((const f32x4*)(eout + ((size_t)e0 * CAP + p0) * DIM))[h];
    if (p1 < CAP) b = ((const f32x4*)(eout + ((size_t)e1 * CAP + p1) * DIM))[h];
    f32x4 o;
    o[0] = g0 * a[0] + g1 * b[0];
    o[1] = g0 * a[1] + g1 * b[1];
    o[2] = g0 * a[2] + g1 * b[2];
    o[3] = g0 * a[3] + g1 * b[3];
    ((f32x4*)(out + (size_t)n * DIM))[h] = o;
}

// ---------------------------------------------------------------------------
extern "C" void kernel_launch(void* const* d_in, const int* in_sizes, int n_in,
                              void* d_out, int out_size, void* d_ws, size_t ws_size,
                              hipStream_t stream) {
    (void)in_sizes; (void)n_in; (void)out_size; (void)ws_size;
    const float* x  = (const float*)d_in[0];   // [N, D]
    const float* Wr = (const float*)d_in[1];   // [D, E]
    const float* br = (const float*)d_in[2];   // [E]
    const float* We = (const float*)d_in[3];   // [E, D, D]
    float* out = (float*)d_out;                // [N, D]

    // workspace layout (all 4-byte elements)
    char* w = (char*)d_ws;
    size_t off = 0;
    float* logits     = (float*)(w + off); off += (size_t)N_TOK * NE * 4;       // 4 MB
    int*   topk       = (int*)  (w + off); off += (size_t)N_TOK * 2 * 4;
    float* gates      = (float*)(w + off); off += (size_t)N_TOK * 2 * 4;
    int*   position   = (int*)  (w + off); off += (size_t)N_TOK * 2 * 4;
    int*   slot_token = (int*)  (w + off); off += (size_t)NE * CAP * 4;
    int*   counts     = (int*)  (w + off); off += 256;
    float* eout       = (float*)(w + off);                                      // 168 MB

    k_routing_gemm<<<dim3(N_TOK / 128), 256, 0, stream>>>(x, Wr, logits);
    k_top2<<<dim3(N_TOK / 256), 256, 0, stream>>>(logits, br, topk, gates);
    k_init_slots<<<dim3((NE * CAP + 255) / 256), 256, 0, stream>>>(slot_token);
    k_positions<<<dim3(NE), 256, 0, stream>>>(topk, position, slot_token, counts);
    k_expert_gemm<<<dim3(DIM / 128, CAP / 128, NE), 256, 0, stream>>>(x, We, slot_token, counts, eout);
    k_combine<<<dim3(N_TOK), 256, 0, stream>>>(eout, topk, gates, position, out);
}
